// WindowAttention_61890478736099
// MI455X (gfx1250) — matrix operand order repaired and checked by reading, not compiled, not run on hardware
//
#include <hip/hip_runtime.h>
#include <hip/hip_bf16.h>

// ---------------------------------------------------------------------------
// CDNA5 (gfx1250) fused Swin window-attention forward.
// One workgroup (8 wave32) per window; bf16 WMMA (v_wmma_f32_16x16x32_bf16),
// f32 accumulate; all intermediates in LDS (208 KB of the 320 KB WGP budget).
// HBM traffic = x once in, out once out; weights/bias L2-resident.
// ---------------------------------------------------------------------------

typedef __attribute__((ext_vector_type(16))) __bf16 v16bf;
typedef __attribute__((ext_vector_type(8)))  float  v8f;

struct B32x2 { uint4 lo, hi; };   // 32 bytes == one 16-element bf16 fragment

__device__ __forceinline__ unsigned short f2bf(float f) {
  unsigned u = __builtin_bit_cast(unsigned, f);
  u += 0x7fffu + ((u >> 16) & 1u);          // round-to-nearest-even
  return (unsigned short)(u >> 16);
}

__device__ __forceinline__ v16bf ld_frag(const void* p0, const void* p1) {
  B32x2 t;
  t.lo = *(const uint4*)p0;                 // b128 load
  t.hi = *(const uint4*)p1;                 // b128 load
  return __builtin_bit_cast(v16bf, t);
}

__device__ __forceinline__ v8f wmma_bf16(v16bf a, v16bf b, v8f c) {
  return __builtin_amdgcn_wmma_f32_16x16x32_bf16(
      false, a, false, b, (short)0, c, false, false);
}

// ---- LDS layout (bytes) ----------------------------------------------------
#define LDS_X   0u        // 64 x 384 bf16 : x (stage 0/1) then attn-out (4/5)
#define LDS_QK  49152u    // 64 x 768 bf16 : Q cols 0-383 | K cols 384-767
#define LDS_VT  147456u   // 384 x 64 bf16 : V transposed (channel-major)
#define LDS_P   196608u   // 8 waves x (16 x 64 bf16) softmax tiles
#define LDS_TOT 212992u

__global__ __launch_bounds__(256)
void swin_attn_kernel(const float* __restrict__ x,      // [2048,49,384]
                      const float* __restrict__ mask,   // [16,49,49]
                      const float* __restrict__ qkv_b,  // [1152]
                      const float* __restrict__ proj_b, // [384]
                      const unsigned short* __restrict__ wqkv,  // bf16 [1152,384]
                      const unsigned short* __restrict__ wproj, // bf16 [384,384]
                      const float* __restrict__ biasbuf,        // [12,49,49]
                      float* __restrict__ out)          // [2048,49,384]
{
  __shared__ __align__(16) unsigned char smem[LDS_TOT];
  const int tid  = threadIdx.x;
  const int wave = tid >> 5;
  const int lane = tid & 31;
  const int ln   = lane & 15;       // position within 16-lane half
  const int hi   = lane >> 4;       // which half of the wave
  const int win  = blockIdx.x;

  // ---- Stage 0: x tile -> LDS bf16 (vectorized), zero-pad rows 49..63 ------
  {
    const float* xw = x + (size_t)win * (49 * 384);
    // 64*384 = 24576 bf16 elements in groups of 4 (49*384 = 18816 = 4704*4)
    for (int i4 = tid; i4 < 6144; i4 += 256) {
      const int i = i4 * 4;
      float4 f;
      if (i4 < 4704) {
        f = *(const float4*)(xw + i);         // global_load_b128
      } else {
        f.x = 0.f; f.y = 0.f; f.z = 0.f; f.w = 0.f;
      }
      uint2 pk;
      pk.x = (unsigned)f2bf(f.x) | ((unsigned)f2bf(f.y) << 16);
      pk.y = (unsigned)f2bf(f.z) | ((unsigned)f2bf(f.w) << 16);
      *(uint2*)(smem + LDS_X + (size_t)i * 2) = pk;   // ds_store_b64
    }
  }
  __syncthreads();

  // ---- Stage 1: qkv = x @ Wqkv^T + b (M=64, N=1152, K=384) -----------------
  // wave owns n-tiles {wave, wave+8, ...}: 9 tiles of 16 output columns
  for (int t = 0; t < 9; ++t) {
    const int nt   = wave + t * 8;       // 0..71
    const int ncol = nt * 16 + ln;       // 0..1151
    v8f acc[4] = {};
    for (int ks = 0; ks < 12; ++ks) {
      // B fragment: B[k][n] = Wqkv[ncol][ks*32 + k]; k contiguous per lane
      const unsigned short* wp = wqkv + ((size_t)ncol * 384 + ks * 32 + hi * 16);
      v16bf bfr = ld_frag(wp, wp + 8);
#pragma unroll
      for (int mt = 0; mt < 4; ++mt) {
        const unsigned char* ap =
            smem + LDS_X + ((mt * 16 + ln) * 384 + ks * 32) * 2 + hi * 32;
        v16bf afr = ld_frag(ap, ap + 16);
        acc[mt] = wmma_bf16(afr, bfr, acc[mt]);
      }
    }
    const float bias = qkv_b[ncol];
    if (nt < 48) {                       // Q (scaled) and K: token-major store
      const float sc = (nt < 24) ? 0.17677669529663687f : 1.0f; // 32^-0.5
#pragma unroll
      for (int mt = 0; mt < 4; ++mt)
#pragma unroll
        for (int v = 0; v < 8; ++v) {
          const int m = mt * 16 + hi * 8 + v;
          *(unsigned short*)(smem + LDS_QK + ((size_t)m * 768 + ncol) * 2) =
              f2bf((acc[mt][v] + bias) * sc);
        }
    } else {                             // V: channel-major (transposed) store
      const int c = ncol - 768;          // 0..383
#pragma unroll
      for (int mt = 0; mt < 4; ++mt) {
        uint4 pk;
        pk.x = (unsigned)f2bf(acc[mt][0] + bias) | ((unsigned)f2bf(acc[mt][1] + bias) << 16);
        pk.y = (unsigned)f2bf(acc[mt][2] + bias) | ((unsigned)f2bf(acc[mt][3] + bias) << 16);
        pk.z = (unsigned)f2bf(acc[mt][4] + bias) | ((unsigned)f2bf(acc[mt][5] + bias) << 16);
        pk.w = (unsigned)f2bf(acc[mt][6] + bias) | ((unsigned)f2bf(acc[mt][7] + bias) << 16);
        *(uint4*)(smem + LDS_VT + ((size_t)c * 64 + mt * 16 + hi * 8) * 2) = pk;
      }
    }
  }
  __syncthreads();

  // ---- Stages 2-4: per-head attention (S in registers, softmax via shfl) ---
  unsigned char* pb = smem + LDS_P + wave * 2048;     // 16x64 bf16 P tile
  for (int round = 0; round < 2; ++round) {
    const int h = wave + round * 8;                   // wave-uniform
    if (h >= 12) continue;
    const float* brow = biasbuf + h * 2401;
    const float* mrow = mask + (size_t)(win & 15) * 2401;
    for (int mt = 0; mt < 4; ++mt) {
      // S(mt, :) = Q(mt) @ K^T   (K = hd = 32 -> one WMMA per tile)
      const unsigned char* qp =
          smem + LDS_QK + ((mt * 16 + ln) * 768 + h * 32) * 2 + hi * 32;
      v16bf qfr = ld_frag(qp, qp + 16);
      v8f s[4];
#pragma unroll
      for (int ntl = 0; ntl < 4; ++ntl) {
        const unsigned char* kp =
            smem + LDS_QK + ((ntl * 16 + ln) * 768 + 384 + h * 32) * 2 + hi * 32;
        v16bf kfr = ld_frag(kp, kp + 16);
        v8f z = {};
        s[ntl] = wmma_bf16(qfr, kfr, z);
      }
      // softmax over the 8 rows held by this half (row m = mt*16 + hi*8 + v)
#pragma unroll
      for (int v = 0; v < 8; ++v) {
        const int m  = mt * 16 + hi * 8 + v;
        const int mr = (m < 49) ? m : 48;   // keep bias/mask reads in-bounds
        float sv[4];
#pragma unroll
        for (int ntl = 0; ntl < 4; ++ntl) {
          const int j = ntl * 16 + ln;
          sv[ntl] = (j < 49)
                        ? (s[ntl][v] + brow[mr * 49 + j] + mrow[mr * 49 + j])
                        : -1e30f;           // mask out K-padding columns
        }
        float mx = fmaxf(fmaxf(sv[0], sv[1]), fmaxf(sv[2], sv[3]));
        mx = fmaxf(mx, __shfl_xor(mx, 1, 32));
        mx = fmaxf(mx, __shfl_xor(mx, 2, 32));
        mx = fmaxf(mx, __shfl_xor(mx, 4, 32));
        mx = fmaxf(mx, __shfl_xor(mx, 8, 32));   // row lives in one 16-lane half
        float sum = 0.f;
#pragma unroll
        for (int ntl = 0; ntl < 4; ++ntl) { sv[ntl] = __expf(sv[ntl] - mx); sum += sv[ntl]; }
        sum += __shfl_xor(sum, 1, 32);
        sum += __shfl_xor(sum, 2, 32);
        sum += __shfl_xor(sum, 4, 32);
        sum += __shfl_xor(sum, 8, 32);
        const float rinv = 1.0f / sum;      // sum >= 1, safe
#pragma unroll
        for (int ntl = 0; ntl < 4; ++ntl) {
          const int j = ntl * 16 + ln;      // padded cols write exact 0
          *(unsigned short*)(pb + ((hi * 8 + v) * 64 + j) * 2) = f2bf(sv[ntl] * rinv);
        }
      }
      // O(mt, :) = P(mt) @ V_h   (K = 64 tokens -> 2 WMMA k-steps)
#pragma unroll
      for (int n2 = 0; n2 < 2; ++n2) {
        const int d = h * 32 + n2 * 16 + ln;          // output channel
        v8f o = {};
#pragma unroll
        for (int kt = 0; kt < 2; ++kt) {
          const unsigned char* pp = pb + (ln * 64 + kt * 32) * 2 + hi * 32;
          v16bf pfr = ld_frag(pp, pp + 16);
          const unsigned char* vp =
              smem + LDS_VT + ((size_t)d * 64 + kt * 32) * 2 + hi * 32;
          v16bf vfr = ld_frag(vp, vp + 16);
          o = wmma_bf16(pfr, vfr, o);
        }
#pragma unroll
        for (int v = 0; v < 8; ++v) {
          const int m = mt * 16 + hi * 8 + v;
          *(unsigned short*)(smem + LDS_X + ((size_t)m * 384 + d) * 2) = f2bf(o[v]);
        }
      }
    }
  }
  __syncthreads();

  // ---- Stage 5: out = attn_out @ Wproj^T + b (M=64, N=384, K=384) ----------
  float* op = out + (size_t)win * 49 * 384;
  for (int t = 0; t < 3; ++t) {
    const int nt   = wave + t * 8;      // 0..23
    const int ncol = nt * 16 + ln;      // 0..383
    v8f acc[4] = {};
    for (int ks = 0; ks < 12; ++ks) {
      const unsigned short* wp = wproj + ((size_t)ncol * 384 + ks * 32 + hi * 16);
      v16bf bfr = ld_frag(wp, wp + 8);
#pragma unroll
      for (int mt = 0; mt < 4; ++mt) {
        const unsigned char* ap =
            smem + LDS_X + ((mt * 16 + ln) * 384 + ks * 32) * 2 + hi * 32;
        v16bf afr = ld_frag(ap, ap + 16);
        acc[mt] = wmma_bf16(afr, bfr, acc[mt]);
      }
    }
    const float bias = proj_b[ncol];
#pragma unroll
    for (int mt = 0; mt < 4; ++mt)
#pragma unroll
      for (int v = 0; v < 8; ++v) {
        const int m = mt * 16 + hi * 8 + v;
        if (m < 49) op[(size_t)m * 384 + ncol] = acc[mt][v] + bias;
      }
  }
}

// ---------------------------------------------------------------------------
// Prep: fp32 weights -> bf16 in workspace; gather relative-position bias into
// a dense [12][49][49] f32 table (both L2-resident for the main kernel).
// ---------------------------------------------------------------------------
__global__ void swin_prep_kernel(const float* __restrict__ qkv_w,
                                 const float* __restrict__ proj_w,
                                 const float* __restrict__ table,
                                 const int*   __restrict__ rel_index,
                                 unsigned short* __restrict__ wqkv,
                                 unsigned short* __restrict__ wproj,
                                 float* __restrict__ biasbuf)
{
  const int i = blockIdx.x * 256 + threadIdx.x;
  if (i < 1152 * 384) wqkv[i]  = f2bf(qkv_w[i]);
  if (i < 384 * 384)  wproj[i] = f2bf(proj_w[i]);
  if (i < 12 * 2401) {
    const int h = i / 2401, ij = i - h * 2401;
    biasbuf[i] = table[rel_index[ij] * 12 + h];
  }
}

extern "C" void kernel_launch(void* const* d_in, const int* in_sizes, int n_in,
                              void* d_out, int out_size, void* d_ws, size_t ws_size,
                              hipStream_t stream) {
  (void)in_sizes; (void)n_in; (void)out_size; (void)ws_size;
  const float* x      = (const float*)d_in[0];
  const float* mask   = (const float*)d_in[1];
  const float* qkv_w  = (const float*)d_in[2];
  const float* qkv_b  = (const float*)d_in[3];
  const float* proj_w = (const float*)d_in[4];
  const float* proj_b = (const float*)d_in[5];
  const float* table  = (const float*)d_in[6];
  const int*   relidx = (const int*)d_in[7];

  // workspace layout (16B aligned): bf16 qkv_w | bf16 proj_w | f32 bias table
  unsigned short* wqkv  = (unsigned short*)d_ws;                          // 884736 B
  unsigned short* wproj = (unsigned short*)((char*)d_ws + 884736);        // 294912 B
  float*          biasb = (float*)((char*)d_ws + 884736 + 294912);        // 115248 B

  swin_prep_kernel<<<(1152 * 384 + 255) / 256, 256, 0, stream>>>(
      qkv_w, proj_w, table, relidx, wqkv, wproj, biasb);
  swin_attn_kernel<<<2048, 256, 0, stream>>>(
      x, mask, qkv_b, proj_b, wqkv, wproj, biasb, (float*)d_out);
}
